// DecoderBlock_69166153335556
// MI455X (gfx1250) — matrix operand order repaired and checked by reading, not compiled, not run on hardware
//
#include <hip/hip_runtime.h>
#include <hip/hip_bf16.h>

// Problem constants (from reference): B=2, S=2048, D=1024, H=16, HD=64, F=4096
#define B_  2
#define S_  2048
#define D_  1024
#define H_  16
#define HD_ 64
#define F_  4096

typedef __bf16 bf16_t;
typedef __attribute__((ext_vector_type(2)))  bf16_t bf16x2;
typedef __attribute__((ext_vector_type(16))) bf16_t bf16x16;
typedef __attribute__((ext_vector_type(8)))  float  f32x8;

// fp32 -> bf16 via native converts (RNE); packed two-at-a-time (v_cvt_pk_bf16_f32)
__device__ __forceinline__ unsigned cvt2(float lo, float hi) {
    bf16x2 t;
    t.x = (bf16_t)lo;
    t.y = (bf16_t)hi;
    return __builtin_bit_cast(unsigned, t);
}

__device__ __forceinline__ unsigned short f2bfu(float f) {
    bf16_t t = (bf16_t)f;
    return __builtin_bit_cast(unsigned short, t);
}

__device__ __forceinline__ uint4 packbf(float4 a, float4 b) {
    uint4 p;
    p.x = cvt2(a.x, a.y);
    p.y = cvt2(a.z, a.w);
    p.z = cvt2(b.x, b.y);
    p.w = cvt2(b.z, b.w);
    return p;
}

union FragBF {
    uint4   q[2];   // 32 bytes
    bf16x16 v;      // 16 bf16 per lane
};

#define BM    128
#define BN    64
#define BK    32
#define PITCH 40   // ushort elems per LDS row: 32 + 8 pad (80B rows, 16B aligned)

// C[M,N] = act( alpha * A[M,K] @ (TRANSB ? Bg^T : Bg)[K,N] + bias + causal )
// fp32 inputs converted to bf16 while staging into LDS; fp32 accumulate.
// 2-stage pipeline: global loads for tile i+1 issue before WMMAs of tile i.
// Batched over blockIdx.z with offset = (z/batchMod)*s1 + (z%batchMod)*s2.
template<bool TRANSB, bool CAUSAL, bool GELU_ACT>
__global__ __launch_bounds__(256) void gemm_bf16_kernel(
    const float* __restrict__ A, const float* __restrict__ Bm,
    const float* __restrict__ bias, float* __restrict__ C,
    int M, int N, int K, int lda, int ldb, int ldc, float alpha,
    int batchMod,
    long aS1, long aS2, long bS1, long bS2, long cS1, long cS2)
{
    __shared__ unsigned short sA[2][BM * PITCH];  // [m][k], k contiguous
    __shared__ unsigned short sB[2][BN * PITCH];  // [n][k], k contiguous

    const int z  = blockIdx.z;
    const int zb = z / batchMod;
    const int zh = z % batchMod;
    A  += zb * aS1 + zh * aS2;
    Bm += zb * bS1 + zh * bS2;
    C  += zb * cS1 + zh * cS2;

    const int tid  = threadIdx.x;
    const int m0   = blockIdx.y * BM;
    const int n0   = blockIdx.x * BN;
    const int wave = tid >> 5;        // 8 waves (wave32)
    const int lane = tid & 31;
    const int lr   = lane & 15;
    const int lh   = lane >> 4;       // lane half selects K sub-blocks
    const int wr   = wave * 16;       // wave row strip within block tile

    f32x8 acc[4] = {};

    // staged global data (regs) for the next K-tile
    float4 ra0, ra1, ra2, ra3, rb0, rb1;

    auto gload = [&](int kk0) {
        // A tile: 128 rows x 32 K; each thread: 16 consecutive floats of one row
        const int r = tid >> 1, half = tid & 1;
        const float4* srcA = (const float4*)(A + (long)(m0 + r) * lda + kk0 + half * 16);
        ra0 = srcA[0]; ra1 = srcA[1]; ra2 = srcA[2]; ra3 = srcA[3];
        if (TRANSB) {
            // Bg is [N, K] row-major: direct copy of 8 floats along K
            const int n = tid >> 2, seg = tid & 3;
            const float4* srcB = (const float4*)(Bm + (long)(n0 + n) * ldb + kk0 + seg * 8);
            rb0 = srcB[0]; rb1 = srcB[1];
        } else {
            // Bg is [K, N] row-major: 8 floats along N (transposed on store)
            const int kk = tid >> 3, seg = tid & 7;
            const float4* srcB = (const float4*)(Bm + (long)(kk0 + kk) * ldb + n0 + seg * 8);
            rb0 = srcB[0]; rb1 = srcB[1];
        }
    };

    auto lstore = [&](unsigned short* dA, unsigned short* dB) {
        const int r = tid >> 1, half = tid & 1;
        *(uint4*)&dA[r * PITCH + half * 16]     = packbf(ra0, ra1);
        *(uint4*)&dA[r * PITCH + half * 16 + 8] = packbf(ra2, ra3);
        if (TRANSB) {
            const int n = tid >> 2, seg = tid & 3;
            *(uint4*)&dB[n * PITCH + seg * 8] = packbf(rb0, rb1);
        } else {
            const int kk = tid >> 3, seg = tid & 7;
            float e[8] = {rb0.x, rb0.y, rb0.z, rb0.w, rb1.x, rb1.y, rb1.z, rb1.w};
            #pragma unroll
            for (int i = 0; i < 8; ++i)
                dB[(seg * 8 + i) * PITCH + kk] = f2bfu(e[i]);
        }
    };

    // prologue: stage tile 0
    gload(0);
    lstore(sA[0], sB[0]);
    __syncthreads();

    int ph = 0;
    for (int k0 = 0; k0 < K; k0 += BK) {
        const bool hasNext = (k0 + BK) < K;
        if (hasNext) gload(k0 + BK);          // issue global loads early

        const unsigned short* cA = sA[ph];
        const unsigned short* cB = sB[ph];

        // Issue ALL fragment loads before any WMMA so the backend can keep
        // distinct registers and use partial s_wait_dscnt between WMMAs.
        // A 16x32 bf16 frag: lane m=lr; half 0 -> K {0..7,16..23}, half 1 -> K {8..15,24..31}
        FragBF a, b[4];
        const int ar = (wr + lr) * PITCH + lh * 8;
        a.q[0] = *(const uint4*)&cA[ar];
        a.q[1] = *(const uint4*)&cA[ar + 16];
        // B 32x16 bf16 frags: lane n=lr; half 0 -> K {0..7,16..23}, half 1 -> K {8..15,24..31}
        #pragma unroll
        for (int nt = 0; nt < 4; ++nt) {
            const int br = (nt * 16 + lr) * PITCH + lh * 8;
            b[nt].q[0] = *(const uint4*)&cB[br];
            b[nt].q[1] = *(const uint4*)&cB[br + 16];
        }
        #pragma unroll
        for (int nt = 0; nt < 4; ++nt) {
            acc[nt] = __builtin_amdgcn_wmma_f32_16x16x32_bf16(
                          false, a.v, false, b[nt].v, (short)0, acc[nt], false, false);
        }

        if (hasNext) lstore(sA[ph ^ 1], sB[ph ^ 1]);  // convert+store after WMMAs issue
        __syncthreads();
        ph ^= 1;
    }

    // hoisted bias (depends only on column)
    float bv[4];
    #pragma unroll
    for (int nt = 0; nt < 4; ++nt)
        bv[nt] = bias ? bias[n0 + nt * 16 + lr] : 0.f;

    // fused epilogue: alpha, bias, causal mask, exact GELU
    #pragma unroll
    for (int nt = 0; nt < 4; ++nt) {
        const int col = n0 + nt * 16 + lr;
        #pragma unroll
        for (int r = 0; r < 8; ++r) {
            const int row = m0 + wr + r + 8 * lh;  // C VGPR r: half0 -> M=r, half1 -> M=8+r
            float val = acc[nt][r] * alpha + bv[nt];
            if (CAUSAL) val += (col > row) ? -1e9f : 0.f;
            if (GELU_ACT) val = 0.5f * val * (1.f + erff(val * 0.70710678118654752f));
            C[(long)row * ldc + col] = val;
        }
    }
}

// In-place softmax over rows of length S_=2048. One 256-thread block per row.
__global__ __launch_bounds__(256) void softmax2048_kernel(float* __restrict__ P)
{
    const long base = (long)blockIdx.x * S_;
    const int tid = threadIdx.x;
    __shared__ float red[256];
    float v[8];
    float mx = -3.4e38f;
    #pragma unroll
    for (int i = 0; i < 8; ++i) {
        v[i] = P[base + tid + i * 256];
        mx = fmaxf(mx, v[i]);
    }
    red[tid] = mx;
    __syncthreads();
    for (int s = 128; s > 0; s >>= 1) {
        if (tid < s) red[tid] = fmaxf(red[tid], red[tid + s]);
        __syncthreads();
    }
    mx = red[0];
    __syncthreads();
    float sum = 0.f;
    #pragma unroll
    for (int i = 0; i < 8; ++i) { v[i] = expf(v[i] - mx); sum += v[i]; }
    red[tid] = sum;
    __syncthreads();
    for (int s = 128; s > 0; s >>= 1) {
        if (tid < s) red[tid] += red[tid + s];
        __syncthreads();
    }
    const float inv = 1.f / red[0];
    #pragma unroll
    for (int i = 0; i < 8; ++i) P[base + tid + i * 256] = v[i] * inv;
}

// out = LayerNorm(X + Y) * g + b, row length D_=1024. One block per row.
__global__ __launch_bounds__(256) void add_layernorm_kernel(
    const float* __restrict__ X, const float* __restrict__ Y,
    const float* __restrict__ g, const float* __restrict__ bta,
    float* __restrict__ out)
{
    const long base = (long)blockIdx.x * D_;
    const int tid = threadIdx.x;
    __shared__ float red[256];
    float v[4];
    float s = 0.f, s2 = 0.f;
    #pragma unroll
    for (int i = 0; i < 4; ++i) {
        const int c = tid + i * 256;
        v[i] = X[base + c] + Y[base + c];
        s  += v[i];
        s2 += v[i] * v[i];
    }
    red[tid] = s;
    __syncthreads();
    for (int t = 128; t > 0; t >>= 1) {
        if (tid < t) red[tid] += red[tid + t];
        __syncthreads();
    }
    s = red[0];
    __syncthreads();
    red[tid] = s2;
    __syncthreads();
    for (int t = 128; t > 0; t >>= 1) {
        if (tid < t) red[tid] += red[tid + t];
        __syncthreads();
    }
    s2 = red[0];
    const float mean = s * (1.f / D_);
    const float var  = s2 * (1.f / D_) - mean * mean;
    const float rstd = rsqrtf(var + 1e-5f);
    #pragma unroll
    for (int i = 0; i < 4; ++i) {
        const int c = tid + i * 256;
        out[base + c] = (v[i] - mean) * rstd * g[c] + bta[c];
    }
}

extern "C" void kernel_launch(void* const* d_in, const int* in_sizes, int n_in,
                              void* d_out, int out_size, void* d_ws, size_t ws_size,
                              hipStream_t stream)
{
    (void)in_sizes; (void)n_in; (void)out_size; (void)ws_size;
    const float* x    = (const float*)d_in[0];
    /* d_in[1] = causal_mask: applied analytically in the scores epilogue */
    const float* wq   = (const float*)d_in[2];
    const float* bq   = (const float*)d_in[3];
    const float* wk   = (const float*)d_in[4];
    const float* bk   = (const float*)d_in[5];
    const float* wv   = (const float*)d_in[6];
    const float* bv   = (const float*)d_in[7];
    const float* wo   = (const float*)d_in[8];
    const float* bo   = (const float*)d_in[9];
    const float* ln1g = (const float*)d_in[10];
    const float* ln1b = (const float*)d_in[11];
    const float* w1   = (const float*)d_in[12];
    const float* b1   = (const float*)d_in[13];
    const float* w2   = (const float*)d_in[14];
    const float* b2   = (const float*)d_in[15];
    const float* ln2g = (const float*)d_in[16];
    const float* ln2b = (const float*)d_in[17];

    float* out   = (float*)d_out;                      // [B,S,D]
    float* attnW = out + (long)B_ * S_ * D_;           // [B,H,S,S] (2nd tuple output)

    // workspace layout (floats), SD = S*D = 2,097,152; total 12*SD = 96 MiB
    const long SD = (long)S_ * D_;
    float* ws = (float*)d_ws;
    float* q  = ws;             // [B,S,D]
    float* k  = ws + 2 * SD;    // [B,S,D]
    float* v  = ws + 4 * SD;    // [B,S,D]
    float* at = ws + 6 * SD;    // attn_out pre-proj [B,S,D]
    float* pj = ws + 8 * SD;    // o-proj out; later reused for ff2 out
    float* h  = ws + 10 * SD;   // LN1 out
    float* ff = ws;             // FFN mid [B,S,F] = 8*SD, aliases dead q/k/v/at

    const int BS = B_ * S_;     // 4096 rows
    dim3 blk(256);

    // 1-3) Q/K/V projections: [BS,D] @ [D,D] + bias
    {
        dim3 g(D_ / BN, BS / BM, 1);
        gemm_bf16_kernel<false, false, false><<<g, blk, 0, stream>>>(
            x, wq, bq, q, BS, D_, D_, D_, D_, D_, 1.f, 1, 0, 0, 0, 0, 0, 0);
        gemm_bf16_kernel<false, false, false><<<g, blk, 0, stream>>>(
            x, wk, bk, k, BS, D_, D_, D_, D_, D_, 1.f, 1, 0, 0, 0, 0, 0, 0);
        gemm_bf16_kernel<false, false, false><<<g, blk, 0, stream>>>(
            x, wv, bv, v, BS, D_, D_, D_, D_, D_, 1.f, 1, 0, 0, 0, 0, 0, 0);
    }

    // 4) scores = Q @ K^T / sqrt(HD) + causal, per (b,h): written straight to d_out
    {
        dim3 g(S_ / BN, S_ / BM, B_ * H_);
        gemm_bf16_kernel<true, true, false><<<g, blk, 0, stream>>>(
            q, k, nullptr, attnW,
            S_, S_, HD_, D_, D_, S_, 0.125f, H_,
            SD, 64, SD, 64, (long)H_ * S_ * S_, (long)S_ * S_);
    }

    // 5) softmax in place on attn_weights
    softmax2048_kernel<<<dim3(B_ * H_ * S_), blk, 0, stream>>>(attnW);

    // 6) attn_out = P @ V, per (b,h); P read fp32 -> bf16 on the fly
    {
        dim3 g(HD_ / BN, S_ / BM, B_ * H_);
        gemm_bf16_kernel<false, false, false><<<g, blk, 0, stream>>>(
            attnW, v, nullptr, at,
            S_, HD_, S_, S_, D_, D_, 1.f, H_,
            (long)H_ * S_ * S_, (long)S_ * S_, SD, 64, SD, 64);
    }

    // 7) output projection
    {
        dim3 g(D_ / BN, BS / BM, 1);
        gemm_bf16_kernel<false, false, false><<<g, blk, 0, stream>>>(
            at, wo, bo, pj, BS, D_, D_, D_, D_, D_, 1.f, 1, 0, 0, 0, 0, 0, 0);
    }

    // 8) h = LN1(x + attn_proj)
    add_layernorm_kernel<<<dim3(BS), blk, 0, stream>>>(x, pj, ln1g, ln1b, h);

    // 9) ff = gelu(h @ w1 + b1)   [BS, F]
    {
        dim3 g(F_ / BN, BS / BM, 1);
        gemm_bf16_kernel<false, false, true><<<g, blk, 0, stream>>>(
            h, w1, b1, ff, BS, F_, D_, D_, F_, F_, 1.f, 1, 0, 0, 0, 0, 0, 0);
    }

    // 10) ff2 = ff @ w2 + b2      [BS, D] (pj reused)
    {
        dim3 g(D_ / BN, BS / BM, 1);
        gemm_bf16_kernel<false, false, false><<<g, blk, 0, stream>>>(
            ff, w2, b2, pj, BS, D_, F_, F_, D_, D_, 1.f, 1, 0, 0, 0, 0, 0, 0);
    }

    // 11) out = LN2(h + ff2)
    add_layernorm_kernel<<<dim3(BS), blk, 0, stream>>>(h, pj, ln2g, ln2b, out);
}
